// Decoder_81235011437021
// MI455X (gfx1250) — matrix operand order repaired and checked by reading, not compiled, not run on hardware
//
#include <hip/hip_runtime.h>

#define B_      32
#define T_ENC   512
#define D_      512
#define N_      200
#define PRENET_ 256
#define QUERY_  1024
#define DEC_    1024
#define ATTN_   128
#define LOC_F_  32
#define LOC_K_  31
#define FRAME_  80
#define ROUT_   160            // R * FRAME_DIM
#define K1_     1792           // prenet(256) + ctx(512) + h1(1024)
#define K2_     2560           // h1(1024) + ctx(512) + h2(1024)
#define G4_     4096           // 4 * hidden
#define KC_     512            // K-chunk staged in LDS (32 rows x 512 x bf16 = 32KB)

typedef __attribute__((ext_vector_type(16))) __bf16 v16bf;
typedef __attribute__((ext_vector_type(8)))  float  v8f;

union ABu { v16bf v; uint4 q[2]; };

__device__ __host__ inline unsigned short f2bf(float f) {
  union { float f; unsigned u; } v; v.f = f;
  unsigned r = v.u + 0x7FFFu + ((v.u >> 16) & 1u);   // round-to-nearest-even
  return (unsigned short)(r >> 16);
}

// ---- WMMA tile loaders matching CDNA5 16-bit A(16x32)/B(32x16) VGPR layouts ----
// A: lanes 0-15 row M=l hold K = k0..k0+7 , k0+16..k0+23 ; lanes 16-31 hold +8.
__device__ inline v16bf loadA(const unsigned short* base, int K, int m0, int k0, int lane) {
  int row = m0 + (lane & 15);
  int hi  = (lane >> 4) << 3;            // 0 or 8
  const unsigned short* p0 = base + (size_t)row * K + k0 + hi;
  ABu u;
  u.q[0] = *(const uint4*)(p0);          // K: k0+hi .. k0+hi+7
  u.q[1] = *(const uint4*)(p0 + 16);     // K: k0+16+hi .. k0+23+hi
  return u.v;
}
// B (weights pre-transposed to Wt[N][K]): lanes 0-15 col n hold K=k0..k0+7, k0+16..k0+23;
// lanes 16-31 hold +8 (same as A).
__device__ inline v16bf loadB(const unsigned short* wt, int K, int n0, int k0, int lane) {
  int col  = n0 + (lane & 15);
  int half = (lane >> 4) << 3;           // 0 or 8
  const unsigned short* p0 = wt + (size_t)col * K + k0 + half;
  ABu u;
  u.q[0] = *(const uint4*)(p0);
  u.q[1] = *(const uint4*)(p0 + 16);
  return u.v;
}

// ---- grid-wide barrier (persistent kernel), sense via generation counter ----
__device__ inline void gbar(unsigned* bar, unsigned nb) {
  __syncthreads();
  if (threadIdx.x == 0) {
    __threadfence();
    unsigned gen = __atomic_load_n(bar + 1, __ATOMIC_RELAXED);
    if (atomicAdd(bar, 1u) + 1u == nb) {
      __atomic_store_n(bar, 0u, __ATOMIC_RELAXED);
      __threadfence();
      atomicAdd(bar + 1, 1u);
    } else {
      while (__atomic_load_n(bar + 1, __ATOMIC_RELAXED) == gen)
        __builtin_amdgcn_s_sleep(2);
    }
    __threadfence();
  }
  __syncthreads();
}

struct Bufs {
  unsigned short *W1t, *W2t, *PREbf;     // prepped (bf16)
  float *H, *PM;                         // prenet hidden, processed memory
  unsigned short *x1, *x2;               // bf16 activation rows [B][K1], [B][K2]
  float *h1, *c1, *h2, *c2, *ctx, *a_cum, *a_old, *alg;
  unsigned *bar;
  float *Z1, *Z2, *fl, *e, *q, *frame;   // scratch
};
struct Ins {
  const float *memory, *frames, *pw1, *pw2, *arn_wx, *arn_wh, *arn_b,
              *drn_wx, *drn_wh, *drn_b, *q_w, *m_w, *loc_k, *loc_w, *attn_v,
              *proj_w, *proj_b, *stop_w, *stop_b;
};

// ---------------- one-time prep kernels ----------------
__global__ void k_zero(unsigned* p, size_t nwords) {
  size_t st = (size_t)gridDim.x * blockDim.x;
  for (size_t i = (size_t)blockIdx.x * blockDim.x + threadIdx.x; i < nwords; i += st)
    p[i] = 0u;
}

// Wt[n][k] = k<kx ? wx[k][n] : wh[k-kx][n]   (stacked, transposed, bf16)
__global__ void k_prep_w(const float* wx, const float* wh, int kx, int ktot,
                         unsigned short* wt) {
  size_t total = (size_t)G4_ * ktot;
  size_t st = (size_t)gridDim.x * blockDim.x;
  for (size_t i = (size_t)blockIdx.x * blockDim.x + threadIdx.x; i < total; i += st) {
    int n = (int)(i / ktot), k = (int)(i % ktot);
    float v = (k < kx) ? wx[(size_t)k * G4_ + n] : wh[(size_t)(k - kx) * G4_ + n];
    wt[i] = f2bf(v);
  }
}

__global__ void k_prenet1(const float* frames, const float* w1, float* H) {
  size_t total = (size_t)B_ * N_ * PRENET_;
  size_t st = (size_t)gridDim.x * blockDim.x;
  for (size_t i = (size_t)blockIdx.x * blockDim.x + threadIdx.x; i < total; i += st) {
    int u = (int)(i % PRENET_);
    size_t bn = i / PRENET_;
    int n = (int)(bn % N_), bb = (int)(bn / N_);
    float s = 0.f;
    if (n > 0) {                                    // shifted: frame -1 is zeros
      const float* fr = frames + ((size_t)bb * N_ + (n - 1)) * FRAME_;
      for (int d = 0; d < FRAME_; ++d) s += fr[d] * w1[d * PRENET_ + u];
    }
    H[i] = fmaxf(s, 0.f);
  }
}

__global__ void k_prenet2(const float* H, const float* w2, unsigned short* PREbf) {
  size_t total = (size_t)B_ * N_ * PRENET_;
  size_t st = (size_t)gridDim.x * blockDim.x;
  for (size_t i = (size_t)blockIdx.x * blockDim.x + threadIdx.x; i < total; i += st) {
    int u = (int)(i % PRENET_);
    size_t bn = i / PRENET_;
    const float* h = H + bn * PRENET_;
    float s = 0.f;
    for (int v = 0; v < PRENET_; ++v) s += h[v] * w2[v * PRENET_ + u];
    PREbf[i] = f2bf(fmaxf(s, 0.f));
  }
}

__global__ void k_procmem(const float* memory, const float* m_w, float* PM) {
  size_t total = (size_t)B_ * T_ENC * ATTN_;
  size_t st = (size_t)gridDim.x * blockDim.x;
  for (size_t i = (size_t)blockIdx.x * blockDim.x + threadIdx.x; i < total; i += st) {
    int j = (int)(i % ATTN_);
    size_t bt = i / ATTN_;
    const float* m = memory + bt * D_;
    float s = 0.f;
    for (int d = 0; d < D_; ++d) s += m[d] * m_w[d * ATTN_ + j];
    PM[i] = s;
  }
}

// ---------------- persistent decoder: all 200 steps ----------------
// Grid: 32 WGs x 256 threads = 256 wave32 waves; one wave owns one N-tile of
// the gate GEMMs, computing both M-tiles (batch halves) so each weight B-tile
// is loaded from L2 once; activations are staged per-WG in 32KB LDS chunks.
__global__ __launch_bounds__(256, 1) void k_decoder(Bufs b, Ins in, float* out) {
  const int NTH  = gridDim.x * blockDim.x;   // 32*256 = 8192
  const int tid  = blockIdx.x * blockDim.x + threadIdx.x;
  const int lane = threadIdx.x & 31;         // wave32
  const int wv   = tid >> 5;                 // 0..255: N-tile owned by this wave
  __shared__ unsigned short ldsx[B_ * KC_];  // 32KB activation chunk
  __shared__ float red[256];

  float* outF = out;                          // [B,N,160]
  float* outS = out + (size_t)B_ * N_ * ROUT_;// [B,N]
  float* outA = outS + (size_t)B_ * N_;       // [B,N,T]

  for (int n = 0; n < N_; ++n) {
    // P0: current prenet slice -> x1 cols [0,256)
    for (int i = tid; i < B_ * PRENET_; i += NTH) {
      int bb = i >> 8, u = i & 255;
      b.x1[bb * K1_ + u] = b.PREbf[((size_t)bb * N_ + n) * PRENET_ + u];
    }
    gbar(b.bar, gridDim.x);

    // P1: attention-LSTM gates  Z1[32][4096] = x1 @ [arn_wx;arn_wh] + b
    {
      const int nt = wv;                       // exactly 256 tiles / 256 waves
      v8f c0 = {0.f,0.f,0.f,0.f,0.f,0.f,0.f,0.f};
      v8f c1 = {0.f,0.f,0.f,0.f,0.f,0.f,0.f,0.f};
      for (int kc0 = 0; kc0 < K1_; kc0 += KC_) {
        int kc = (K1_ - kc0 < KC_) ? (K1_ - kc0) : KC_;
        // cooperative stage of x1[:, kc0:kc0+kc) into LDS (b128 copies)
        for (int idx = threadIdx.x; idx < B_ * (kc >> 3); idx += blockDim.x) {
          int row = idx / (kc >> 3), seg = idx % (kc >> 3);
          ((uint4*)ldsx)[row * (KC_ >> 3) + seg] =
              *(const uint4*)(b.x1 + (size_t)row * K1_ + kc0 + seg * 8);
        }
        __syncthreads();
        for (int k0 = 0; k0 < kc; k0 += 32) {
          v16bf a0 = loadA(ldsx, KC_, 0,  k0, lane);
          v16bf a1 = loadA(ldsx, KC_, 16, k0, lane);
          v16bf bm = loadB(b.W1t, K1_, nt * 16, kc0 + k0, lane);
          c0 = __builtin_amdgcn_wmma_f32_16x16x32_bf16(false, a0, false, bm,
                                                       (short)0, c0, false, false);
          c1 = __builtin_amdgcn_wmma_f32_16x16x32_bf16(false, a1, false, bm,
                                                       (short)0, c1, false, false);
        }
        __syncthreads();
      }
      int col = nt * 16 + (lane & 15);
      float bias = in.arn_b[col];
      int r0 = (lane >> 4) * 8;
      #pragma unroll
      for (int r = 0; r < 8; ++r) {
        b.Z1[(r0 + r) * G4_ + col]      = c0[r] + bias;
        b.Z1[(16 + r0 + r) * G4_ + col] = c1[r] + bias;
      }
    }
    gbar(b.bar, gridDim.x);

    // P2: LSTM1 elementwise -> h1,c1 ; stash h1 bf16 for next-step x1 and this-step x2
    for (int i = tid; i < B_ * QUERY_; i += NTH) {
      int bb = i >> 10, u = i & 1023;
      const float* z = b.Z1 + (size_t)bb * G4_;
      float ig = 1.f / (1.f + __expf(-z[u]));
      float fg = 1.f / (1.f + __expf(-z[u + 1024]));
      float gg = tanhf(z[u + 2048]);
      float og = 1.f / (1.f + __expf(-z[u + 3072]));
      float cn = fg * b.c1[i] + ig * gg;
      float hn = og * tanhf(cn);
      b.c1[i] = cn; b.h1[i] = hn;
      unsigned short hb = f2bf(hn);
      b.x1[bb * K1_ + 768 + u] = hb;
      b.x2[bb * K2_ + u]       = hb;
    }
    gbar(b.bar, gridDim.x);

    // P3: q = h1 @ q_w ; fl = location conv over (a_cum, a_old) [pre-update]
    for (int i = tid; i < B_ * ATTN_; i += NTH) {
      int bb = i >> 7, j = i & 127;
      const float* h = b.h1 + (size_t)bb * QUERY_;
      float s = 0.f;
      for (int u = 0; u < QUERY_; ++u) s += h[u] * in.q_w[u * ATTN_ + j];
      b.q[i] = s;
    }
    for (int i = tid; i < B_ * T_ENC * LOC_F_; i += NTH) {
      int f = i & 31, t = (i >> 5) & 511, bb = i >> 14;
      const float* ac = b.a_cum + (size_t)bb * T_ENC;
      const float* ao = b.a_old + (size_t)bb * T_ENC;
      float s = 0.f;
      for (int k = 0; k < LOC_K_; ++k) {
        int tt = t + k - 15;                       // SAME padding
        if (tt < 0 || tt >= T_ENC) continue;
        s += ac[tt] * in.loc_k[(k * 2 + 0) * LOC_F_ + f]
           + ao[tt] * in.loc_k[(k * 2 + 1) * LOC_F_ + f];
      }
      b.fl[i] = s;
    }
    gbar(b.bar, gridDim.x);

    // P4: energies e[b,t] = tanh(q + fl@loc_w + proc_mem) . attn_v
    for (int i = tid; i < B_ * T_ENC; i += NTH) {
      int bb = i >> 9;
      const float* flv = b.fl + (size_t)i * LOC_F_;
      const float* pm  = b.PM + (size_t)i * ATTN_;
      const float* qv  = b.q  + (size_t)bb * ATTN_;
      float acc = 0.f;
      for (int j = 0; j < ATTN_; ++j) {
        float s = qv[j] + pm[j];
        for (int f = 0; f < LOC_F_; ++f) s += flv[f] * in.loc_w[f * ATTN_ + j];
        acc += tanhf(s) * in.attn_v[j];
      }
      b.e[i] = acc;
    }
    gbar(b.bar, gridDim.x);

    // P5: softmax per batch row (one block per row); update a_cum/a_old; emit attention
    {
      int bb = blockIdx.x;                       // 32 blocks == 32 rows
      float* ev = b.e + (size_t)bb * T_ENC;
      float m = -1e30f;
      for (int t = threadIdx.x; t < T_ENC; t += blockDim.x) m = fmaxf(m, ev[t]);
      red[threadIdx.x] = m; __syncthreads();
      for (int s = 128; s > 0; s >>= 1) {
        if ((int)threadIdx.x < s) red[threadIdx.x] = fmaxf(red[threadIdx.x], red[threadIdx.x + s]);
        __syncthreads();
      }
      m = red[0]; __syncthreads();
      float sum = 0.f;
      for (int t = threadIdx.x; t < T_ENC; t += blockDim.x) {
        float x = __expf(ev[t] - m); ev[t] = x; sum += x;
      }
      red[threadIdx.x] = sum; __syncthreads();
      for (int s = 128; s > 0; s >>= 1) {
        if ((int)threadIdx.x < s) red[threadIdx.x] += red[threadIdx.x + s];
        __syncthreads();
      }
      float inv = 1.f / red[0]; __syncthreads();
      for (int t = threadIdx.x; t < T_ENC; t += blockDim.x) {
        float al = ev[t] * inv;
        b.alg[bb * T_ENC + t]    = al;
        b.a_old[bb * T_ENC + t]  = al;
        b.a_cum[bb * T_ENC + t] += al;
        outA[((size_t)bb * N_ + n) * T_ENC + t] = al;
      }
    }
    gbar(b.bar, gridDim.x);

    // P6: context = align @ memory ; bf16 into x1 cols[256..768) and x2 cols[1024..1536)
    for (int i = tid; i < B_ * D_; i += NTH) {
      int bb = i >> 9, d = i & 511;
      const float* al  = b.alg + (size_t)bb * T_ENC;
      const float* mem = in.memory + (size_t)bb * T_ENC * D_ + d;
      float s = 0.f;
      for (int t = 0; t < T_ENC; ++t) s += al[t] * mem[(size_t)t * D_];
      b.ctx[i] = s;
      unsigned short cb = f2bf(s);
      b.x1[bb * K1_ + 256 + d]  = cb;
      b.x2[bb * K2_ + 1024 + d] = cb;
    }
    gbar(b.bar, gridDim.x);

    // P7: decoder-LSTM gates  Z2[32][4096] = x2 @ [drn_wx;drn_wh] + b
    {
      const int nt = wv;
      v8f c0 = {0.f,0.f,0.f,0.f,0.f,0.f,0.f,0.f};
      v8f c1 = {0.f,0.f,0.f,0.f,0.f,0.f,0.f,0.f};
      for (int kc0 = 0; kc0 < K2_; kc0 += KC_) {
        int kc = (K2_ - kc0 < KC_) ? (K2_ - kc0) : KC_;
        for (int idx = threadIdx.x; idx < B_ * (kc >> 3); idx += blockDim.x) {
          int row = idx / (kc >> 3), seg = idx % (kc >> 3);
          ((uint4*)ldsx)[row * (KC_ >> 3) + seg] =
              *(const uint4*)(b.x2 + (size_t)row * K2_ + kc0 + seg * 8);
        }
        __syncthreads();
        for (int k0 = 0; k0 < kc; k0 += 32) {
          v16bf a0 = loadA(ldsx, KC_, 0,  k0, lane);
          v16bf a1 = loadA(ldsx, KC_, 16, k0, lane);
          v16bf bm = loadB(b.W2t, K2_, nt * 16, kc0 + k0, lane);
          c0 = __builtin_amdgcn_wmma_f32_16x16x32_bf16(false, a0, false, bm,
                                                       (short)0, c0, false, false);
          c1 = __builtin_amdgcn_wmma_f32_16x16x32_bf16(false, a1, false, bm,
                                                       (short)0, c1, false, false);
        }
        __syncthreads();
      }
      int col = nt * 16 + (lane & 15);
      float bias = in.drn_b[col];
      int r0 = (lane >> 4) * 8;
      #pragma unroll
      for (int r = 0; r < 8; ++r) {
        b.Z2[(r0 + r) * G4_ + col]      = c0[r] + bias;
        b.Z2[(16 + r0 + r) * G4_ + col] = c1[r] + bias;
      }
    }
    gbar(b.bar, gridDim.x);

    // P8: LSTM2 elementwise -> h2,c2 ; stash h2 bf16 for next-step x2
    for (int i = tid; i < B_ * DEC_; i += NTH) {
      int bb = i >> 10, u = i & 1023;
      const float* z = b.Z2 + (size_t)bb * G4_;
      float ig = 1.f / (1.f + __expf(-z[u]));
      float fg = 1.f / (1.f + __expf(-z[u + 1024]));
      float gg = tanhf(z[u + 2048]);
      float og = 1.f / (1.f + __expf(-z[u + 3072]));
      float cn = fg * b.c2[i] + ig * gg;
      float hn = og * tanhf(cn);
      b.c2[i] = cn; b.h2[i] = hn;
      b.x2[bb * K2_ + 1536 + u] = f2bf(hn);
    }
    gbar(b.bar, gridDim.x);

    // P9: frame = [h2;ctx] @ proj_w + proj_b  -> out + scratch for stopnet
    for (int i = tid; i < B_ * ROUT_; i += NTH) {
      int bb = i / ROUT_, o = i - bb * ROUT_;
      float s = in.proj_b[o];
      const float* h = b.h2 + (size_t)bb * DEC_;
      for (int j = 0; j < DEC_; ++j) s += h[j] * in.proj_w[j * ROUT_ + o];
      const float* cx = b.ctx + (size_t)bb * D_;
      for (int j = 0; j < D_; ++j) s += cx[j] * in.proj_w[(DEC_ + j) * ROUT_ + o];
      b.frame[i] = s;
      outF[((size_t)bb * N_ + n) * ROUT_ + o] = s;
    }
    gbar(b.bar, gridDim.x);

    // P10: stop = [h2;frame] @ stop_w + stop_b
    for (int i = tid; i < B_; i += NTH) {
      float s = in.stop_b[0];
      const float* h = b.h2 + (size_t)i * DEC_;
      for (int j = 0; j < DEC_; ++j) s += h[j] * in.stop_w[j];
      const float* fr = b.frame + (size_t)i * ROUT_;
      for (int o = 0; o < ROUT_; ++o) s += fr[o] * in.stop_w[DEC_ + o];
      outS[(size_t)i * N_ + n] = s;
    }
    gbar(b.bar, gridDim.x);
  }
}

extern "C" void kernel_launch(void* const* d_in, const int* in_sizes, int n_in,
                              void* d_out, int out_size, void* d_ws, size_t ws_size,
                              hipStream_t stream) {
  (void)in_sizes; (void)n_in; (void)out_size; (void)ws_size;
  Ins in;
  in.memory = (const float*)d_in[0];
  in.frames = (const float*)d_in[1];
  in.pw1    = (const float*)d_in[2];
  in.pw2    = (const float*)d_in[3];
  in.arn_wx = (const float*)d_in[4];
  in.arn_wh = (const float*)d_in[5];
  in.arn_b  = (const float*)d_in[6];
  in.drn_wx = (const float*)d_in[7];
  in.drn_wh = (const float*)d_in[8];
  in.drn_b  = (const float*)d_in[9];
  in.q_w    = (const float*)d_in[10];
  in.m_w    = (const float*)d_in[11];
  in.loc_k  = (const float*)d_in[12];
  in.loc_w  = (const float*)d_in[13];
  in.attn_v = (const float*)d_in[14];
  in.proj_w = (const float*)d_in[15];
  in.proj_b = (const float*)d_in[16];
  in.stop_w = (const float*)d_in[17];
  in.stop_b = (const float*)d_in[18];

  char* p = (char*)d_ws;
  auto take = [&](size_t bytes) -> char* {
    char* r = p; p += (bytes + 255) & ~(size_t)255; return r;
  };

  Bufs b;
  b.W1t   = (unsigned short*)take((size_t)G4_ * K1_ * 2);
  b.W2t   = (unsigned short*)take((size_t)G4_ * K2_ * 2);
  b.PREbf = (unsigned short*)take((size_t)B_ * N_ * PRENET_ * 2);
  b.H     = (float*)take((size_t)B_ * N_ * PRENET_ * 4);
  b.PM    = (float*)take((size_t)B_ * T_ENC * ATTN_ * 4);
  char* state0 = p;                         // zero-initialized region
  b.x1    = (unsigned short*)take((size_t)B_ * K1_ * 2);
  b.x2    = (unsigned short*)take((size_t)B_ * K2_ * 2);
  b.h1    = (float*)take((size_t)B_ * QUERY_ * 4);
  b.c1    = (float*)take((size_t)B_ * QUERY_ * 4);
  b.h2    = (float*)take((size_t)B_ * DEC_ * 4);
  b.c2    = (float*)take((size_t)B_ * DEC_ * 4);
  b.ctx   = (float*)take((size_t)B_ * D_ * 4);
  b.a_cum = (float*)take((size_t)B_ * T_ENC * 4);
  b.a_old = (float*)take((size_t)B_ * T_ENC * 4);
  b.alg   = (float*)take((size_t)B_ * T_ENC * 4);
  b.bar   = (unsigned*)take(256);
  size_t stateWords = (size_t)(p - state0) / 4;
  b.Z1    = (float*)take((size_t)B_ * G4_ * 4);
  b.Z2    = (float*)take((size_t)B_ * G4_ * 4);
  b.fl    = (float*)take((size_t)B_ * T_ENC * LOC_F_ * 4);
  b.e     = (float*)take((size_t)B_ * T_ENC * 4);
  b.q     = (float*)take((size_t)B_ * ATTN_ * 4);
  b.frame = (float*)take((size_t)B_ * ROUT_ * 4);

  // one-time prep (stream-ordered)
  k_zero   <<<256,  256, 0, stream>>>((unsigned*)state0, stateWords);
  k_prep_w <<<2048, 256, 0, stream>>>(in.arn_wx, in.arn_wh, 768,  K1_, b.W1t);
  k_prep_w <<<2048, 256, 0, stream>>>(in.drn_wx, in.drn_wh, 1536, K2_, b.W2t);
  k_prenet1<<<2048, 256, 0, stream>>>(in.frames, in.pw1, b.H);
  k_prenet2<<<2048, 256, 0, stream>>>(b.H, in.pw2, b.PREbf);
  k_procmem<<<2048, 256, 0, stream>>>(in.memory, in.m_w, b.PM);

  // persistent decoder: 32 WGs x 256 threads = 256 wave32 waves
  k_decoder<<<32, 256, 0, stream>>>(b, in, (float*)d_out);
}
